// _HamburgerNDPlus_69226282877036
// MI455X (gfx1250) — matrix operand order repaired and checked by reading, not compiled, not run on hardware
//
#include <hip/hip_runtime.h>
#include <hip/hip_bf16.h>

// ---------------------------------------------------------------------------
// MI455X (gfx1250) implementation of the HamburgerNDPlus reference.
// All heavy GEMMs run through v_wmma_f32_16x16x32_bf16 (bf16 in, f32 acc),
// double-buffered LDS software pipeline, async global->LDS staging for the
// A tile (GLOBAL_LOAD_ASYNC_TO_LDS_B128 / ASYNCcnt) when the toolchain
// exposes it.
// ---------------------------------------------------------------------------

typedef __attribute__((ext_vector_type(16))) __bf16 v16bf;
typedef __attribute__((ext_vector_type(8)))  float  v8f;
typedef __attribute__((ext_vector_type(4)))  int    v4i;

#define LDSP 40   // padded LDS row stride (elements); 80B rows keep 16B chunk alignment

#define ASG __attribute__((address_space(1)))
#define ASL __attribute__((address_space(3)))

#if __has_builtin(__builtin_amdgcn_global_load_async_to_lds_b128)
#define USE_ASYNC_LDS 1
#else
#define USE_ASYNC_LDS 0
#endif

#if USE_ASYNC_LDS
#if __has_builtin(__builtin_amdgcn_s_wait_asynccnt)
#define WAIT_ASYNC() __builtin_amdgcn_s_wait_asynccnt(0)
#else
#define WAIT_ASYNC() asm volatile("s_wait_asynccnt 0" ::: "memory")
#endif
#else
#define WAIT_ASYNC() ((void)0)
#endif

union AFragU { v16bf v; uint4 u[2]; };

// ---- epilogue parameter bundle --------------------------------------------
struct EpiP {
    const float* bias;
    const float* gamma; const float* beta; const float* mean; const float* var;
    const float* zres;  const float* c_ham; const float* c_sc;
    float*  outf;
    __bf16* outb;
    long out_stride;   // per-grid.z-unit stride of output (elements)
    long z_stride;     // per-unit stride of zres (elements)
    int  ldOut;
};

// ---- A tile staging: straight copy, async memory->LDS when available ------
template<int BM>
__device__ __forceinline__ void stage_A(__bf16* sA, const __bf16* __restrict__ A,
                                        long ldA, int m0, int kk, int K, int tid) {
    constexpr int PER = (BM * 32 / 8) / 256;  // 8-elem (16B) chunks per thread
#pragma unroll
    for (int i = 0; i < PER; ++i) {
        int ci  = tid + i * 256;
        int row = ci >> 2;            // 4 chunks per 32-wide row
        int c8  = (ci & 3) * 8;
        const __bf16* src = A + (long)(m0 + row) * ldA + kk + c8;
        __bf16* dst = sA + row * LDSP + c8;
#if USE_ASYNC_LDS
        __builtin_amdgcn_global_load_async_to_lds_b128((ASG v4i*)src, (ASL v4i*)dst, 0, 0);
#else
        *(uint4*)dst = *(const uint4*)src;
#endif
        if (kk + 32 < K) __builtin_prefetch((const void*)(src + 32), 0, 1);
    }
}

// ---- B tile: global->regs (issued early), regs->LDS transposed (late) -----
template<int BN, bool BF32>
__device__ __forceinline__ void load_B_regs(uint4* q, const void* __restrict__ Bv,
                                            long ldB, int n0, int kk, int K, int tid) {
    constexpr int CPR = BN / 8;
    constexpr int PER = (32 * CPR) / 256;
#pragma unroll
    for (int i = 0; i < PER; ++i) {
        int ci = tid + i * 256;
        int kr = ci / CPR;
        int c8 = (ci % CPR) * 8;
        if constexpr (BF32) {
            const uint4* src = (const uint4*)((const float*)Bv + (long)(kk + kr) * ldB + n0 + c8);
            q[2 * i]     = src[0];
            q[2 * i + 1] = src[1];
            if (kk + 32 < K)
                __builtin_prefetch((const void*)((const float*)Bv + (long)(kk + 32 + kr) * ldB + n0 + c8), 0, 1);
        } else {
            const __bf16* src = (const __bf16*)Bv + (long)(kk + kr) * ldB + n0 + c8;
            q[i] = *(const uint4*)src;
            if (kk + 32 < K) __builtin_prefetch((const void*)(src + 32 * ldB), 0, 1);
        }
    }
}

template<int BN, bool BF32>
__device__ __forceinline__ void store_B_lds(__bf16* sBt, const uint4* q, int tid) {
    constexpr int CPR = BN / 8;
    constexpr int PER = (32 * CPR) / 256;
#pragma unroll
    for (int i = 0; i < PER; ++i) {
        int ci = tid + i * 256;
        int kr = ci / CPR;
        int c8 = (ci % CPR) * 8;
        if constexpr (BF32) {
            const float* f = (const float*)&q[2 * i];
#pragma unroll
            for (int j = 0; j < 8; ++j) sBt[(c8 + j) * LDSP + kr] = (__bf16)f[j];
        } else {
            const __bf16* p = (const __bf16*)&q[i];
#pragma unroll
            for (int j = 0; j < 8; ++j) sBt[(c8 + j) * LDSP + kr] = p[j];
        }
    }
}

// ---- fragment loads matching the CDNA5 16-bit WMMA VGPR layouts -----------
__device__ __forceinline__ v16bf load_frag_A(const __bf16* sA, int rowBase, int lane) {
    int hi = (lane >> 4) & 1, lo = lane & 15;
    AFragU f;
    const __bf16* p = sA + (rowBase + lo) * LDSP + hi * 8;
    f.u[0] = *(const uint4*)p;
    f.u[1] = *(const uint4*)(p + 16);
    return f.v;
}
__device__ __forceinline__ v16bf load_frag_B(const __bf16* sBt, int colBase, int lane) {
    int hi = (lane >> 4) & 1, lo = lane & 15;
    AFragU f;
    const __bf16* p = sBt + (colBase + lo) * LDSP + hi * 8;
    f.u[0] = *(const uint4*)p;
    f.u[1] = *(const uint4*)(p + 16);
    return f.v;
}

// ---------------------------------------------------------------------------
// Generic bf16 WMMA GEMM: OUT(M x N) = A(M x K, row-major bf16) * B(K x N, row-major)
// Double-buffered LDS, one barrier per K-step, next tile's global traffic
// issued before the WMMAs of the current tile.
// EPI: 0 raw f32, 1 bias+relu->bf16, 2 ->bf16, 3 BN+relu->bf16,
//      4 bias+BN + coef mix with residual z + relu -> f32.
// ---------------------------------------------------------------------------
template<int BM, int BN, int WMv, int WNv, int EPI, bool BF32>
__global__ __launch_bounds__(256) void gemm_k(const __bf16* __restrict__ A,
                                              const void* __restrict__ Bv,
                                              int K, long ldA, long ldB,
                                              long strA, long strB, EpiP ep) {
    constexpr int WTM = BM / WMv, WTN = BN / WNv;
    constexpr int MF = WTM / 16, NF = WTN / 16;
    constexpr int NQ = ((32 * (BN / 8)) / 256) * (BF32 ? 2 : 1);
    __shared__ __bf16 sA[2][BM * LDSP];
    __shared__ __bf16 sBt[2][BN * LDSP];

    const int tid = threadIdx.x, lane = tid & 31, wave = tid >> 5;
    const int wm = wave / WNv, wn = wave % WNv;
    const int m0 = blockIdx.y * BM, n0 = blockIdx.x * BN, u = blockIdx.z;

    const __bf16* Ab = A + (long)u * strA;
    const void* Bu;
    if constexpr (BF32) Bu = (const void*)((const float*)Bv + (long)u * strB);
    else                Bu = (const void*)((const __bf16*)Bv + (long)u * strB);

    v8f acc[MF][NF];
#pragma unroll
    for (int mt = 0; mt < MF; ++mt)
#pragma unroll
        for (int nt = 0; nt < NF; ++nt)
#pragma unroll
            for (int v = 0; v < 8; ++v) acc[mt][nt][v] = 0.f;

    uint4 breg[NQ];

    // prologue: stage tile 0
    stage_A<BM>(sA[0], Ab, ldA, m0, 0, K, tid);
    load_B_regs<BN, BF32>(breg, Bu, ldB, n0, 0, K, tid);
    store_B_lds<BN, BF32>(sBt[0], breg, tid);
    WAIT_ASYNC();
    __syncthreads();

    for (int kk = 0; kk < K; kk += 32) {
        const int cur = (kk >> 5) & 1, nxt = cur ^ 1;
        v16bf af[MF], bfr[NF];
#pragma unroll
        for (int mt = 0; mt < MF; ++mt) af[mt] = load_frag_A(sA[cur], wm * WTM + mt * 16, lane);
#pragma unroll
        for (int nt = 0; nt < NF; ++nt) bfr[nt] = load_frag_B(sBt[cur], wn * WTN + nt * 16, lane);

        const bool more = (kk + 32) < K;
        if (more) {
            load_B_regs<BN, BF32>(breg, Bu, ldB, n0, kk + 32, K, tid);  // global latency ...
            stage_A<BM>(sA[nxt], Ab, ldA, m0, kk + 32, K, tid);         // ... hides under WMMAs
        }
#pragma unroll
        for (int mt = 0; mt < MF; ++mt)
#pragma unroll
            for (int nt = 0; nt < NF; ++nt)
                acc[mt][nt] = __builtin_amdgcn_wmma_f32_16x16x32_bf16(
                    false, af[mt], false, bfr[nt], (short)0, acc[mt][nt], false, false);
        if (more) store_B_lds<BN, BF32>(sBt[nxt], breg, tid);
        WAIT_ASYNC();
        __syncthreads();
    }

    float cham = 0.f, csc = 0.f;
    if constexpr (EPI == 4) { cham = *ep.c_ham; csc = *ep.c_sc; }
    const int hi = (lane >> 4) & 1, lo = lane & 15;
#pragma unroll
    for (int mt = 0; mt < MF; ++mt)
#pragma unroll
        for (int nt = 0; nt < NF; ++nt) {
            const int col = n0 + wn * WTN + nt * 16 + lo;
#pragma unroll
            for (int v = 0; v < 8; ++v) {
                const int row = m0 + wm * WTM + mt * 16 + hi * 8 + v;
                float val = acc[mt][nt][v];
                const long oidx = (long)u * ep.out_stride + (long)row * ep.ldOut + col;
                if constexpr (EPI == 0) {
                    ep.outf[oidx] = val;
                } else if constexpr (EPI == 1) {
                    val += ep.bias[row];
                    ep.outb[oidx] = (__bf16)fmaxf(val, 0.f);
                } else if constexpr (EPI == 2) {
                    ep.outb[oidx] = (__bf16)val;
                } else if constexpr (EPI == 3) {
                    float inv = ep.gamma[row] * rsqrtf(ep.var[row] + 1e-5f);
                    val = val * inv + (ep.beta[row] - ep.mean[row] * inv);
                    ep.outb[oidx] = (__bf16)fmaxf(val, 0.f);
                } else {  // EPI == 4
                    val += ep.bias[row];
                    float inv = ep.gamma[row] * rsqrtf(ep.var[row] + 1e-5f);
                    float h = val * inv + (ep.beta[row] - ep.mean[row] * inv);
                    float o = cham * h +
                              csc * ep.zres[(long)u * ep.z_stride + (long)row * 4096 + col];
                    ep.outf[oidx] = fmaxf(o, 0.f);
                }
            }
        }
}

// ---------------------------------------------------------------------------
// cos = Dn^T * X (64 x 4096 per unit) with fused column softmax over r=64.
// BM=64 covers all r; lane pair (L, L^16) holds a full column -> softmax via
// __shfl_xor(.,16) on wave32.  Same double-buffered pipeline as gemm_k.
// ---------------------------------------------------------------------------
__global__ __launch_bounds__(256) void cos_softmax_k(const __bf16* __restrict__ Drd,
                                                     const __bf16* __restrict__ X,
                                                     const float* __restrict__ invn,
                                                     __bf16* __restrict__ Cnr,
                                                     __bf16* __restrict__ Crn) {
    constexpr int BM = 64, BN = 128, MF = 4;
    __shared__ __bf16 sA[2][BM * LDSP];
    __shared__ __bf16 sBt[2][BN * LDSP];

    const int tid = threadIdx.x, lane = tid & 31, wave = tid >> 5;
    const int n0 = blockIdx.x * BN, u = blockIdx.z;
    const __bf16* Ab = Drd + (long)u * 64 * 512;
    const __bf16* Bb = X + (long)u * 512 * 4096;

    v8f acc[MF];
#pragma unroll
    for (int mt = 0; mt < MF; ++mt)
#pragma unroll
        for (int v = 0; v < 8; ++v) acc[mt][v] = 0.f;

    uint4 breg[2];
    stage_A<BM>(sA[0], Ab, 512, 0, 0, 512, tid);
    load_B_regs<BN, false>(breg, (const void*)Bb, 4096, n0, 0, 512, tid);
    store_B_lds<BN, false>(sBt[0], breg, tid);
    WAIT_ASYNC();
    __syncthreads();

    for (int kk = 0; kk < 512; kk += 32) {
        const int cur = (kk >> 5) & 1, nxt = cur ^ 1;
        v16bf af[MF];
#pragma unroll
        for (int mt = 0; mt < MF; ++mt) af[mt] = load_frag_A(sA[cur], mt * 16, lane);
        v16bf bfr = load_frag_B(sBt[cur], wave * 16, lane);

        const bool more = (kk + 32) < 512;
        if (more) {
            load_B_regs<BN, false>(breg, (const void*)Bb, 4096, n0, kk + 32, 512, tid);
            stage_A<BM>(sA[nxt], Ab, 512, 0, kk + 32, 512, tid);
        }
#pragma unroll
        for (int mt = 0; mt < MF; ++mt)
            acc[mt] = __builtin_amdgcn_wmma_f32_16x16x32_bf16(
                false, af[mt], false, bfr, (short)0, acc[mt], false, false);
        if (more) store_B_lds<BN, false>(sBt[nxt], breg, tid);
        WAIT_ASYNC();
        __syncthreads();
    }

    const int hi = (lane >> 4) & 1, lo = lane & 15;
    const int ncol = n0 + wave * 16 + lo;
    const float scale = invn[u * 4096 + ncol] * 0.1f;  // (1/T), T = 10

    float mx = -3.4e38f;
#pragma unroll
    for (int mt = 0; mt < MF; ++mt)
#pragma unroll
        for (int v = 0; v < 8; ++v) {
            float s = acc[mt][v] * scale;
            acc[mt][v] = s;
            mx = fmaxf(mx, s);
        }
    mx = fmaxf(mx, __shfl_xor(mx, 16, 32));

    float sum = 0.f;
#pragma unroll
    for (int mt = 0; mt < MF; ++mt)
#pragma unroll
        for (int v = 0; v < 8; ++v) {
            float e = __expf(acc[mt][v] - mx);
            acc[mt][v] = e;
            sum += e;
        }
    sum += __shfl_xor(sum, 16, 32);
    const float rs = 1.f / sum;

    __bf16* cnr = Cnr + (long)u * 4096 * 64 + (long)ncol * 64;
    __bf16* crn = Crn + (long)u * 64 * 4096 + ncol;
#pragma unroll
    for (int mt = 0; mt < MF; ++mt)
#pragma unroll
        for (int v = 0; v < 8; ++v) {
            int r = mt * 16 + hi * 8 + v;
            __bf16 c = (__bf16)(acc[mt][v] * rs);
            cnr[r] = c;
            crn[(long)r * 4096] = c;
        }
}

// ---------------------------------------------------------------------------
// Small helper kernels
// ---------------------------------------------------------------------------
__global__ void f2bf_k(const float* __restrict__ s, __bf16* __restrict__ d, int n) {
    int i = blockIdx.x * blockDim.x + threadIdx.x;
    if (i < n) d[i] = (__bf16)s[i];
}

__global__ void dinit_k(const float* __restrict__ Ds, const float* __restrict__ Dc,
                        __bf16* __restrict__ Drd) {
    int u = blockIdx.x, r = threadIdx.x;             // 16 units x 64 r
    int b = u >> 1;
    const float* src = ((u & 1) ? Dc : Ds) + (long)b * 512 * 64;
    float ss = 0.f;
    for (int d = 0; d < 512; ++d) { float x = src[(long)d * 64 + r]; ss += x * x; }
    float inv = 1.f / fmaxf(sqrtf(ss), 1e-6f);
    __bf16* o = Drd + (long)u * 64 * 512 + (long)r * 512;
    for (int d = 0; d < 512; ++d) o[d] = (__bf16)(src[(long)d * 64 + r] * inv);
}

__global__ void colinv_k(const __bf16* __restrict__ X, float* __restrict__ invn) {
    int u = blockIdx.y;
    int n = blockIdx.x * blockDim.x + threadIdx.x;   // 4096 columns
    const __bf16* xb = X + (long)u * 512 * 4096 + n;
    float ss = 0.f;
    for (int d = 0; d < 512; ++d) { float x = (float)xb[(long)d * 4096]; ss += x * x; }
    invn[u * 4096 + n] = 1.f / fmaxf(sqrtf(ss), 1e-6f);
}

// Column-normalize XCt over d -> new D in both (r,d) and (d,r) layouts.
// (The reference's division by sums[r] cancels under the subsequent normalize.)
__global__ void dnorm_k(const float* __restrict__ XCt, __bf16* __restrict__ Drd,
                        __bf16* __restrict__ Ddr) {
    int u = blockIdx.x, r = threadIdx.x;
    const float* s = XCt + (long)u * 512 * 64;
    float ss = 0.f;
    for (int d = 0; d < 512; ++d) { float x = s[(long)d * 64 + r]; ss += x * x; }
    float inv = 1.f / fmaxf(sqrtf(ss), 1e-6f);
    __bf16* ord = Drd + (long)u * 64 * 512 + (long)r * 512;
    __bf16* odr = Ddr + (long)u * 512 * 64 + r;
    for (int d = 0; d < 512; ++d) {
        __bf16 v = (__bf16)(s[(long)d * 64 + r] * inv);
        ord[d] = v;
        odr[(long)d * 64] = v;
    }
}

// ---------------------------------------------------------------------------
extern "C" void kernel_launch(void* const* d_in, const int* in_sizes, int n_in,
                              void* d_out, int out_size, void* d_ws, size_t ws_size,
                              hipStream_t stream) {
    (void)in_sizes; (void)n_in; (void)out_size; (void)ws_size;

    const float* z        = (const float*)d_in[0];   // (8,512,64,64)
    const float* lower_w  = (const float*)d_in[1];   // (1024,512)
    const float* lower_b  = (const float*)d_in[2];   // (1024,)
    const float* Dinit_s  = (const float*)d_in[3];   // (8,512,64)
    const float* Dinit_c  = (const float*)d_in[4];   // (8,512,64)
    const float* cheese_w = (const float*)d_in[5];   // (1024,1024)
    const float* ch_gamma = (const float*)d_in[6];
    const float* ch_beta  = (const float*)d_in[7];
    const float* ch_mean  = (const float*)d_in[8];
    const float* ch_var   = (const float*)d_in[9];
    const float* upper_w  = (const float*)d_in[10];  // (512,1024)
    const float* upper_b  = (const float*)d_in[11];
    const float* h_gamma  = (const float*)d_in[12];
    const float* h_beta   = (const float*)d_in[13];
    const float* h_mean   = (const float*)d_in[14];
    const float* h_var    = (const float*)d_in[15];
    const float* coef_ham = (const float*)d_in[16];
    const float* coef_sc  = (const float*)d_in[17];
    float* out = (float*)d_out;

    // ---- workspace bump allocator (256B aligned) --------------------------
    char* ws = (char*)d_ws;
    size_t off = 0;
    auto alloc = [&](size_t bytes) -> void* {
        void* p = ws + off;
        off += (bytes + 255) & ~(size_t)255;
        return p;
    };
    __bf16* wLow = (__bf16*)alloc((size_t)1024 * 512 * 2);
    __bf16* wCh  = (__bf16*)alloc((size_t)1024 * 1024 * 2);
    __bf16* wUp  = (__bf16*)alloc((size_t)512 * 1024 * 2);
    __bf16* X    = (__bf16*)alloc((size_t)8 * 1024 * 4096 * 2);   // lower output; reused for Xbar
    __bf16* Y2   = (__bf16*)alloc((size_t)8 * 1024 * 4096 * 2);   // cheese output
    float*  invn = (float*) alloc((size_t)16 * 4096 * 4);
    __bf16* Drd  = (__bf16*)alloc((size_t)16 * 64 * 512 * 2);
    __bf16* Ddr  = (__bf16*)alloc((size_t)16 * 512 * 64 * 2);
    __bf16* Cnr  = (__bf16*)alloc((size_t)16 * 4096 * 64 * 2);
    __bf16* Crn  = (__bf16*)alloc((size_t)16 * 64 * 4096 * 2);
    float*  XCt  = (float*) alloc((size_t)16 * 512 * 64 * 4);

    // ---- weight conversion + D init ---------------------------------------
    f2bf_k<<<2048, 256, 0, stream>>>(lower_w,  wLow, 1024 * 512);
    f2bf_k<<<4096, 256, 0, stream>>>(cheese_w, wCh,  1024 * 1024);
    f2bf_k<<<2048, 256, 0, stream>>>(upper_w,  wUp,  512 * 1024);
    dinit_k<<<16, 64, 0, stream>>>(Dinit_s, Dinit_c, Drd);

    // ---- lower conv: X = relu(Wl * z + b), bf16, (B,1024,4096) ------------
    {
        EpiP ep = {};
        ep.bias = lower_b; ep.outb = X;
        ep.out_stride = (long)1024 * 4096; ep.ldOut = 4096;
        gemm_k<128, 128, 2, 4, 1, true><<<dim3(32, 8, 8), 256, 0, stream>>>(
            wLow, (const void*)z, 512, 512, 4096, 0, (long)512 * 4096, ep);
    }

    // ---- per-column inverse norms of X ------------------------------------
    colinv_k<<<dim3(16, 16), 256, 0, stream>>>(X, invn);

    // ---- 3 VQ iterations (16 batch-half units) ----------------------------
    for (int it = 0; it < 3; ++it) {
        cos_softmax_k<<<dim3(32, 1, 16), 256, 0, stream>>>(Drd, X, invn, Cnr, Crn);
        EpiP ep = {};
        ep.outf = XCt; ep.out_stride = (long)512 * 64; ep.ldOut = 64;
        gemm_k<128, 64, 2, 4, 0, false><<<dim3(1, 4, 16), 256, 0, stream>>>(
            X, (const void*)Cnr, 4096, 4096, 64, (long)512 * 4096, (long)4096 * 64, ep);
        dnorm_k<<<16, 64, 0, stream>>>(XCt, Drd, Ddr);
    }

    // ---- Xbar = D * C  -> overwrite X (bf16) ------------------------------
    {
        EpiP ep = {};
        ep.outb = X; ep.out_stride = (long)512 * 4096; ep.ldOut = 4096;
        gemm_k<128, 128, 2, 4, 2, false><<<dim3(32, 4, 16), 256, 0, stream>>>(
            Ddr, (const void*)Crn, 64, 64, 4096, (long)512 * 64, (long)64 * 4096, ep);
    }

    // ---- cheese conv + BN + relu -> Y2 ------------------------------------
    {
        EpiP ep = {};
        ep.gamma = ch_gamma; ep.beta = ch_beta; ep.mean = ch_mean; ep.var = ch_var;
        ep.outb = Y2; ep.out_stride = (long)1024 * 4096; ep.ldOut = 4096;
        gemm_k<128, 128, 2, 4, 3, false><<<dim3(32, 8, 8), 256, 0, stream>>>(
            wCh, (const void*)X, 1024, 1024, 4096, 0, (long)1024 * 4096, ep);
    }

    // ---- upper conv + BN + residual mix + relu -> d_out -------------------
    {
        EpiP ep = {};
        ep.bias = upper_b;
        ep.gamma = h_gamma; ep.beta = h_beta; ep.mean = h_mean; ep.var = h_var;
        ep.zres = z; ep.z_stride = (long)512 * 4096;
        ep.c_ham = coef_ham; ep.c_sc = coef_sc;
        ep.outf = out; ep.out_stride = (long)512 * 4096; ep.ldOut = 4096;
        gemm_k<128, 128, 2, 4, 4, false><<<dim3(32, 4, 8), 256, 0, stream>>>(
            wUp, (const void*)Y2, 1024, 1024, 4096, 0, (long)1024 * 4096, ep);
    }
}
